// ScaledTernary_90434831385362
// MI455X (gfx1250) — matrix operand order repaired and checked by reading, not compiled, not run on hardware
//
#include <hip/hip_runtime.h>

typedef __attribute__((ext_vector_type(16))) _Float16 v16h;
typedef __attribute__((ext_vector_type(8)))  _Float16 v8h;
typedef __attribute__((ext_vector_type(8)))  float    v8f;

#define ROWS   8192
#define DIN    4096
#define UNITS  4096

// ---------------- async global->LDS plumbing (gfx1250) ----------------
#if defined(__HIP_DEVICE_COMPILE__) && __has_builtin(__builtin_amdgcn_global_load_async_to_lds_b128)
#define USE_ASYNC 1
#else
#define USE_ASYNC 0
#endif

typedef int vi4 __attribute__((vector_size(16)));
typedef __attribute__((address_space(1))) vi4* gvi4p;   // global 16B vector ptr
typedef __attribute__((address_space(3))) vi4* lvi4p;   // LDS    16B vector ptr
typedef __attribute__((address_space(1))) void gbl_void;
typedef __attribute__((address_space(3))) void lds_void;

#define GCAST(p) ((gvi4p)(gbl_void*)(void*)(p))
#define LCAST(p) ((lvi4p)(lds_void*)(void*)(p))

#if defined(__HIP_DEVICE_COMPILE__) && __has_builtin(__builtin_amdgcn_s_wait_asynccnt)
#define WAIT_ASYNC0() __builtin_amdgcn_s_wait_asynccnt(0)
#else
#define WAIT_ASYNC0() asm volatile("s_wait_asynccnt 0" ::: "memory")
#endif

__device__ __forceinline__ _Float16 tern(float w) {
    // SteTern(0.5): exact in f16
    return (_Float16)((w > 0.5f) ? 1.0f : ((w < -0.5f) ? -1.0f : 0.0f));
}

// =====================================================================
// Pass A: x (fp32) -> xh (f16), streaming convert
// =====================================================================
__global__ __launch_bounds__(256)
void cvt_x_f16(const float* __restrict__ x, _Float16* __restrict__ xh) {
    size_t i = (size_t)blockIdx.x * 256 + threadIdx.x;   // one float4 each
    float4 v = ((const float4*)x)[i];
    union { _Float16 h[4]; uint2 u; } p;
    p.h[0] = (_Float16)v.x; p.h[1] = (_Float16)v.y;
    p.h[2] = (_Float16)v.z; p.h[3] = (_Float16)v.w;
    ((uint2*)xh)[i] = p.u;
}

// =====================================================================
// Pass B: W (fp32, [k][n]) -> wtT (ternary f16, TRANSPOSED [n][k])
// =====================================================================
#define TT 64
__global__ __launch_bounds__(256)
void tern_transpose(const float* __restrict__ W, _Float16* __restrict__ wtT) {
    __shared__ __attribute__((aligned(16))) _Float16 t[TT][TT + 8];  // row stride 144B
    const int tid = threadIdx.x;
    const int k0 = blockIdx.y * TT;
    const int n0 = blockIdx.x * TT;
#pragma unroll
    for (int i = 0; i < 4; ++i) {
        const int f  = tid + i * 256;
        const int r  = f >> 4;
        const int c4 = f & 15;
        float4 v = ((const float4*)W)[(size_t)(k0 + r) * (UNITS / 4) + (n0 >> 2) + c4];
        t[c4 * 4 + 0][r] = tern(v.x);
        t[c4 * 4 + 1][r] = tern(v.y);
        t[c4 * 4 + 2][r] = tern(v.z);
        t[c4 * 4 + 3][r] = tern(v.w);
    }
    __syncthreads();
    const int n   = tid >> 2;
    const int seg = tid & 3;
    uint4 a = *(const uint4*)&t[n][seg * 16];
    uint4 b = *(const uint4*)&t[n][seg * 16 + 8];
    _Float16* dst = &wtT[(size_t)(n0 + n) * DIN + k0 + seg * 16];
    *(uint4*)(dst)     = a;
    *(uint4*)(dst + 8) = b;
}

// =====================================================================
// Main GEMM on pre-converted f16 data.
// BM=128, BN=128, BK=64; 8 waves 4(M)x2(N); wave tile 32x64 = 2x4 accums.
// Double-buffered LDS, COMPILE-TIME buffer indices (manual unroll by 2).
// =====================================================================
#define BM  128
#define BN  128
#define BK2 64
#define NKT2 (DIN / BK2)   // 64 stages (even)

__global__ __launch_bounds__(256)
void gemm_f16_pre(const _Float16* __restrict__ xh,
                  const _Float16* __restrict__ wtT,
                  const float* __restrict__ scale,
                  const float* __restrict__ bias,
                  float* __restrict__ out)
{
    __shared__ __attribute__((aligned(32))) _Float16 ldsA[2][BM * BK2];  // 16KB x2
    __shared__ __attribute__((aligned(32))) _Float16 ldsB[2][BN * BK2];  // 16KB x2

    const int tid     = threadIdx.x;
    const int lane    = tid & 31;
    const int wave    = tid >> 5;
    const int wm      = (wave & 3) * 32;
    const int wn      = (wave >> 2) * 64;
    const int block_m = blockIdx.y * BM;
    const int block_n = blockIdx.x * BN;

    v8f acc[2][4];
#pragma unroll
    for (int mt = 0; mt < 2; ++mt)
#pragma unroll
        for (int nt = 0; nt < 4; ++nt)
            acc[mt][nt] = {};

    // staging map: f = tid + i*256 (0..1023); row = f>>3 (0..127); cg = f&7 (8 halves)
#if USE_ASYNC
    auto stage_async = [&](int k0, int buf) {
#pragma unroll
        for (int i = 0; i < 4; ++i) {
            const int f = tid + i * 256;
            const int row = f >> 3, cg = f & 7;
            const _Float16* ga = xh  + (size_t)(block_m + row) * DIN + k0 + cg * 8;
            const _Float16* gb = wtT + (size_t)(block_n + row) * DIN + k0 + cg * 8;
            __builtin_amdgcn_global_load_async_to_lds_b128(
                GCAST(ga), LCAST(&ldsA[buf][row * BK2 + cg * 8]), 0, 0);
            __builtin_amdgcn_global_load_async_to_lds_b128(
                GCAST(gb), LCAST(&ldsB[buf][row * BK2 + cg * 8]), 0, 0);
        }
    };
#else
    uint4 aReg[4], bReg[4];
    auto stage_load = [&](int k0) {
#pragma unroll
        for (int i = 0; i < 4; ++i) {
            const int f = tid + i * 256;
            const int row = f >> 3, cg = f & 7;
            aReg[i] = *(const uint4*)(xh  + (size_t)(block_m + row) * DIN + k0 + cg * 8);
            bReg[i] = *(const uint4*)(wtT + (size_t)(block_n + row) * DIN + k0 + cg * 8);
        }
    };
    auto stage_store = [&](int buf) {
#pragma unroll
        for (int i = 0; i < 4; ++i) {
            const int f = tid + i * 256;
            const int row = f >> 3, cg = f & 7;
            *(uint4*)&ldsA[buf][row * BK2 + cg * 8] = aReg[i];
            *(uint4*)&ldsB[buf][row * BK2 + cg * 8] = bReg[i];
        }
    };
#endif

    // compute one K=64 stage out of LDS buffer `buf` (call with constant buf!)
    auto compute_stage = [&](const int buf) {
        const int half = lane & 15;
        const int kbA  = (lane >> 4) << 4;   // 0 or 16 halves
        const int kbB  = (lane >> 4) << 4;   // 0 or 16 halves
#pragma unroll
        for (int j = 0; j < 2; ++j) {        // two K=32 steps
            v16h af[2], bf[4];
#pragma unroll
            for (int mt = 0; mt < 2; ++mt) {
                const int m = wm + mt * 16 + half;
                af[mt] = *(const v16h*)&ldsA[buf][m * BK2 + j * 32 + kbA];
            }
#pragma unroll
            for (int nt = 0; nt < 4; ++nt) {
                const int n = wn + nt * 16 + half;
                bf[nt] = *(const v16h*)&ldsB[buf][n * BK2 + j * 32 + kbB];
            }
#pragma unroll
            for (int mt = 0; mt < 2; ++mt)
#pragma unroll
                for (int nt = 0; nt < 4; ++nt)
                    acc[mt][nt] = __builtin_amdgcn_wmma_f32_16x16x32_f16(
                        false, af[mt], false, bf[nt],
                        (short)0, acc[mt][nt], false, false);
        }
    };

    // ------------- pipelined main loop, manual unroll x2, constant buffers -------------
#if USE_ASYNC
    stage_async(0, 0);
    WAIT_ASYNC0();
    __syncthreads();
#pragma unroll 1
    for (int kt = 0; kt < NKT2; kt += 2) {
        stage_async((kt + 1) * BK2, 1);          // kt+1 <= 63 always valid
        compute_stage(0);
        WAIT_ASYNC0();
        __syncthreads();
        if (kt + 2 < NKT2) stage_async((kt + 2) * BK2, 0);
        compute_stage(1);
        WAIT_ASYNC0();
        __syncthreads();
    }
#else
    stage_load(0);
    stage_store(0);
    __syncthreads();
#pragma unroll 1
    for (int kt = 0; kt < NKT2; kt += 2) {
        stage_load((kt + 1) * BK2);
        compute_stage(0);
        stage_store(1);
        __syncthreads();
        if (kt + 2 < NKT2) stage_load((kt + 2) * BK2);
        compute_stage(1);
        if (kt + 2 < NKT2) stage_store(0);
        __syncthreads();
    }
#endif

    // ---------------- epilogue ----------------
#pragma unroll
    for (int mt = 0; mt < 2; ++mt) {
        const int row0 = block_m + wm + mt * 16 + ((lane >> 4) << 3);
#pragma unroll
        for (int nt = 0; nt < 4; ++nt) {
            const int col = block_n + wn + nt * 16 + (lane & 15);
            const float s = scale[col];
            const float b = bias[col];
#pragma unroll
            for (int v = 0; v < 8; ++v) {
                float y = acc[mt][nt][v] * s + b;
                y = fminf(fmaxf(y, -100.0f), 100.0f);
                __builtin_nontemporal_store(y, &out[(size_t)(row0 + v) * UNITS + col]);
            }
        }
    }
}

// =====================================================================
// Fallback: fused single-pass kernel (for small ws)
// =====================================================================
#define BK 32
#define NKT (DIN / BK)

__global__ __launch_bounds__(256)
void ternary_gemm_wmma(const float* __restrict__ x,
                       const float* __restrict__ W,
                       const float* __restrict__ scale,
                       const float* __restrict__ bias,
                       float* __restrict__ out)
{
    __shared__ __attribute__((aligned(32))) _Float16 ldsA[BM * BK];
    __shared__ __attribute__((aligned(32))) _Float16 ldsB[BN * BK];

    const int tid     = threadIdx.x;
    const int lane    = tid & 31;
    const int wave    = tid >> 5;
    const int wm      = (wave & 3) * 32;
    const int wn      = (wave >> 2) * 64;
    const int block_m = blockIdx.y * BM;
    const int block_n = blockIdx.x * BN;

    v8f acc[2][4];
#pragma unroll
    for (int mt = 0; mt < 2; ++mt)
#pragma unroll
        for (int nt = 0; nt < 4; ++nt)
            acc[mt][nt] = {};

    float4 aReg[4], bReg[4];

    auto load_global = [&](int k0) {
#pragma unroll
        for (int i = 0; i < 4; ++i) {
            const int f = tid + i * 256;
            const int arow = f >> 3, acg = f & 7;
            aReg[i] = ((const float4*)x)[(size_t)(block_m + arow) * (DIN / 4)
                                         + (k0 >> 2) + acg];
            const int krow = f >> 5, bcg = f & 31;
            bReg[i] = ((const float4*)W)[(size_t)(k0 + krow) * (UNITS / 4)
                                         + (block_n >> 2) + bcg];
        }
    };

    auto store_lds = [&]() {
#pragma unroll
        for (int i = 0; i < 4; ++i) {
            const int f = tid + i * 256;
            const int arow = f >> 3, acg = f & 7;
            union { _Float16 h[4]; uint2 u; } p;
            p.h[0] = (_Float16)aReg[i].x;
            p.h[1] = (_Float16)aReg[i].y;
            p.h[2] = (_Float16)aReg[i].z;
            p.h[3] = (_Float16)aReg[i].w;
            *(uint2*)&ldsA[arow * BK + acg * 4] = p.u;
            const int krow = f >> 5, bcg = f & 31;
            ldsB[(bcg * 4 + 0) * BK + krow] = tern(bReg[i].x);
            ldsB[(bcg * 4 + 1) * BK + krow] = tern(bReg[i].y);
            ldsB[(bcg * 4 + 2) * BK + krow] = tern(bReg[i].z);
            ldsB[(bcg * 4 + 3) * BK + krow] = tern(bReg[i].w);
        }
    };

    auto compute = [&]() {
        const int half = lane & 15;
        v16h afrag[2], bfrag[4];
        const int kbA = (lane >> 4) << 4;
#pragma unroll
        for (int mt = 0; mt < 2; ++mt) {
            const int m  = wm + mt * 16 + half;
            afrag[mt] = *(const v16h*)&ldsA[m * BK + kbA];
        }
        const int kbB = (lane >> 4) << 4;
#pragma unroll
        for (int nt = 0; nt < 4; ++nt) {
            const int n = wn + nt * 16 + half;
            bfrag[nt] = *(const v16h*)&ldsB[n * BK + kbB];
        }
#pragma unroll
        for (int mt = 0; mt < 2; ++mt)
#pragma unroll
            for (int nt = 0; nt < 4; ++nt)
                acc[mt][nt] = __builtin_amdgcn_wmma_f32_16x16x32_f16(
                    false, afrag[mt], false, bfrag[nt],
                    (short)0, acc[mt][nt], false, false);
    };

    load_global(0);
    store_lds();
    __syncthreads();

#pragma unroll 1
    for (int kt = 0; kt < NKT; ++kt) {
        if (kt + 1 < NKT) load_global((kt + 1) * BK);
        compute();
        __syncthreads();
        if (kt + 1 < NKT) {
            store_lds();
            __syncthreads();
        }
    }

#pragma unroll
    for (int mt = 0; mt < 2; ++mt) {
        const int row0 = block_m + wm + mt * 16 + ((lane >> 4) << 3);
#pragma unroll
        for (int nt = 0; nt < 4; ++nt) {
            const int col = block_n + wn + nt * 16 + (lane & 15);
            const float s = scale[col];
            const float b = bias[col];
#pragma unroll
            for (int v = 0; v < 8; ++v) {
                float y = acc[mt][nt][v] * s + b;
                y = fminf(fmaxf(y, -100.0f), 100.0f);
                __builtin_nontemporal_store(y, &out[(size_t)(row0 + v) * UNITS + col]);
            }
        }
    }
}

// =====================================================================
extern "C" void kernel_launch(void* const* d_in, const int* in_sizes, int n_in,
                              void* d_out, int out_size, void* d_ws, size_t ws_size,
                              hipStream_t stream) {
    const float* x     = (const float*)d_in[0];
    const float* W     = (const float*)d_in[1];
    const float* scale = (const float*)d_in[2];
    const float* bias  = (const float*)d_in[3];
    float*       out   = (float*)d_out;

    const size_t need = (size_t)ROWS * DIN * 2 + (size_t)DIN * UNITS * 2;  // 96 MB
    if (ws_size >= need) {
        _Float16* xh  = (_Float16*)d_ws;
        _Float16* wtT = xh + (size_t)ROWS * DIN;
        cvt_x_f16<<<(ROWS * (size_t)DIN / 4 + 255) / 256, 256, 0, stream>>>(x, xh);
        tern_transpose<<<dim3(UNITS / TT, DIN / TT), 256, 0, stream>>>(W, wtT);
        gemm_f16_pre<<<dim3(UNITS / BN, ROWS / BM), 256, 0, stream>>>(xh, wtT, scale, bias, out);
    } else {
        ternary_gemm_wmma<<<dim3(UNITS / BN, ROWS / BM), 256, 0, stream>>>(x, W, scale, bias, out);
    }
}
